// SelfAttention_23450521436634
// MI455X (gfx1250) — matrix operand order repaired and checked by reading, not compiled, not run on hardware
//
#include <hip/hip_runtime.h>
#include <hip/hip_bf16.h>

typedef __bf16 bf16;
typedef __attribute__((ext_vector_type(16))) __bf16 v16bf;
typedef __attribute__((ext_vector_type(8)))  __bf16 v8bf;
typedef __attribute__((ext_vector_type(8)))  float  v8f;

// Problem constants
constexpr int Bc  = 2;
constexpr int Sc  = 2048;
constexpr int Dc  = 1024;   // feature dim (also K-dim of all GEMMs)
constexpr int Hc  = 16;
constexpr int HDc = 64;
constexpr int Nc  = 1024;   // H*HD == OUT
constexpr int Mc  = 4096;   // B*S

// 1/sqrt(HD) * log2(e): fold log2e into Q so softmax can use exp2 exactly.
#define QSCALE 0.18033688011112042f

static __device__ __forceinline__ v8f wmma_bf16(v16bf a, v16bf b, v8f c) {
  return __builtin_amdgcn_wmma_f32_16x16x32_bf16(false, a, false, b, (short)0, c,
                                                 false, false);
}

// A-fragment (16x32 bf16): lane (&15)=row M; elems 0-7 -> K = kc+8*half+e,
// elems 8-15 -> K = kc+16+8*half+(e-8). Two 16B contiguous loads per lane.
static __device__ __forceinline__ v16bf load_a_frag(const bf16* row, int kc, int half) {
  union { v16bf v; v8bf h[2]; } u;
  u.h[0] = *(const v8bf*)(row + kc + 8 * half);
  u.h[1] = *(const v8bf*)(row + kc + 16 + 8 * half);
  return u.v;
}

// B-fragment (32x16 bf16): lane (&15)=col N; elems 0-7 -> K = kc+8*half+e,
// elems 8-15 -> K = kc+16+8*half+(e-8). Two 16B contiguous loads per lane (source stored row-per-column, K-contiguous).
static __device__ __forceinline__ v16bf load_b_frag(const bf16* colrow, int kc, int half) {
  union { v16bf v; v8bf h[2]; } u;
  u.h[0] = *(const v8bf*)(colrow + kc + 8 * half);
  u.h[1] = *(const v8bf*)(colrow + kc + 16 + 8 * half);
  return u.v;
}

// ---------------- CDNA5 async / TDM paths ----------------
#if defined(__has_builtin)
#if __has_builtin(__builtin_amdgcn_global_load_async_to_lds_b128)
#define HAVE_ASYNC_LDS 1
#endif
#if __has_builtin(__builtin_amdgcn_tensor_load_to_lds)
#define HAVE_TDM 1
#endif
#endif
#ifndef HAVE_ASYNC_LDS
#define HAVE_ASYNC_LDS 0
#endif
#ifndef HAVE_TDM
#define HAVE_TDM 0
#endif

typedef int v4i __attribute__((vector_size(16)));
typedef __attribute__((address_space(1))) v4i* gv4i_p;   // global int4*
typedef __attribute__((address_space(3))) v4i* lv4i_p;   // LDS int4*

typedef __attribute__((ext_vector_type(4))) unsigned int u32x4;
typedef __attribute__((ext_vector_type(4))) int  i32x4;
typedef __attribute__((ext_vector_type(8))) int  i32x8;

static __device__ __forceinline__ void async_copy16(const bf16* g, bf16* l) {
#if HAVE_ASYNC_LDS
  __builtin_amdgcn_global_load_async_to_lds_b128((gv4i_p)(g), (lv4i_p)(l), 0, 0);
#else
  *(v8bf*)l = *(const v8bf*)g;   // fallback: plain load + ds_store
#endif
}

static __device__ __forceinline__ void async_wait_all() {
#if HAVE_ASYNC_LDS
#if defined(__has_builtin) && __has_builtin(__builtin_amdgcn_s_wait_asynccnt)
  __builtin_amdgcn_s_wait_asynccnt(0);
#else
  asm volatile("s_wait_asynccnt 0x0" ::: "memory");
#endif
#endif
}

// LDS byte offset of a __shared__ object (addrspacecast strips the aperture)
typedef __attribute__((address_space(3))) const void* lds_cvp;
static __device__ __forceinline__ unsigned lds_off(const bf16* p) {
  return (unsigned)(unsigned long long)(lds_cvp)p;
}

// Tensor DMA descriptor group 0: count=1, lds_addr, 57-bit global addr, type=2
static __device__ __forceinline__ u32x4 tdm_group0(unsigned lds_addr,
                                                   unsigned long long gaddr) {
  u32x4 g;
  g.x = 1u;                                              // count=1, user descriptor
  g.y = lds_addr;                                        // LDS byte address
  g.z = (unsigned)(gaddr & 0xffffffffu);                 // global_addr[31:0]
  g.w = (unsigned)((gaddr >> 32) & 0x1ffffffu) | (2u << 30);  // addr[56:32], type=2
  return g;
}

// Tensor DMA descriptor group 1 for a 2-D tile of 2-byte elements.
static __device__ __forceinline__ i32x8 tdm_group1(unsigned tensor_d0, unsigned tensor_d1,
                                                   unsigned tile_d0, unsigned tile_d1,
                                                   unsigned long long d0_stride) {
  i32x8 g;
  g[0] = (int)(1u << 16);                                       // data_size=1 -> 2 bytes
  g[1] = (int)((tensor_d0 & 0xffffu) << 16);                    // tensor_dim0[15:0] @63:48
  g[2] = (int)(((tensor_d0 >> 16) & 0xffffu) |                  // tensor_dim0[31:16]
               ((tensor_d1 & 0xffffu) << 16));                  // tensor_dim1[15:0]
  g[3] = (int)(((tensor_d1 >> 16) & 0xffffu) |                  // tensor_dim1[31:16]
               ((tile_d0 & 0xffffu) << 16));                    // tile_dim0 @127:112
  g[4] = (int)(tile_d1 & 0xffffu);                              // tile_dim1; tile_dim2=0
  g[5] = (int)(unsigned)(d0_stride & 0xffffffffu);              // dim0_stride[31:0]
  g[6] = (int)(unsigned)((d0_stride >> 32) & 0xffffu);          // dim0_stride[47:32]
  g[7] = 0;                                                     // dim1_stride unused (2D)
  return g;
}

// Load a rows x cols 2-byte-element tile (row stride = `stride` elems) into LDS.
static __device__ __forceinline__ void tdm_load_tile(const bf16* gsrc, const bf16* ldst,
                                                     unsigned rows, unsigned cols,
                                                     unsigned long long stride) {
#if HAVE_TDM
  u32x4 g0 = tdm_group0(lds_off(ldst), (unsigned long long)(const void*)gsrc);
  i32x8 g1 = tdm_group1(cols, rows, cols, rows, stride);
  __builtin_amdgcn_tensor_load_to_lds(g0, g1, (i32x4)0, (i32x4)0, (i32x8)0, 0);
#else
  (void)gsrc; (void)ldst; (void)rows; (void)cols; (void)stride;
#endif
}

static __device__ __forceinline__ void tdm_wait_all() {
#if HAVE_TDM
#if defined(__has_builtin) && __has_builtin(__builtin_amdgcn_s_wait_tensorcnt)
  __builtin_amdgcn_s_wait_tensorcnt(0);
#else
  asm volatile("s_wait_tensorcnt 0x0" ::: "memory");
#endif
#endif
}

// xor-swizzle within groups of 32 lanes (masks <16 stay inside each 16-lane half)
template <int MASK>
static __device__ __forceinline__ float swz_xor(float x) {
  int i = __builtin_bit_cast(int, x);
  i = __builtin_amdgcn_ds_swizzle(i, (MASK << 10) | 0x1F);
  return __builtin_bit_cast(float, i);
}
static __device__ __forceinline__ float redmax16(float v) {
  v = fmaxf(v, swz_xor<1>(v));
  v = fmaxf(v, swz_xor<2>(v));
  v = fmaxf(v, swz_xor<4>(v));
  v = fmaxf(v, swz_xor<8>(v));
  return v;
}
static __device__ __forceinline__ float redsum16(float v) {
  v += swz_xor<1>(v);
  v += swz_xor<2>(v);
  v += swz_xor<4>(v);
  v += swz_xor<8>(v);
  return v;
}

// ---------------- cast / transpose helpers ----------------
__global__ void k_cast_bf16(const float* __restrict__ src, bf16* __restrict__ dst, int n) {
  int i = blockIdx.x * blockDim.x + threadIdx.x;
  if (i < n) dst[i] = (bf16)src[i];
}

// dst[n][f] = (bf16) src[f][n], both 1024x1024
__global__ void k_transpose_cast(const float* __restrict__ src, bf16* __restrict__ dst) {
  int idx = blockIdx.x * blockDim.x + threadIdx.x;  // idx = n*1024 + f
  int n = idx >> 10, f = idx & 1023;
  dst[idx] = (bf16)src[f * Dc + n];
}

// ---------------- WMMA GEMM: C[M=4096, N=1024] = A[M,K=1024] * Bt[N,K]^T ----
// mode 0: store Q  [B,H,S,HD] bf16, scaled by QSCALE (+bias)
// mode 1: store K  [B,H,S,HD] bf16 (+bias)
// mode 2: store V^T [B,H,HD,S] bf16 (+bias)
// mode 3: store f32 [M,N] row-major (+bias)
static __device__ __forceinline__ void store_tile(v8f acc, int m0, int n0, int l15,
                                                  int half, const float* __restrict__ bias,
                                                  void* __restrict__ dstv, int mode) {
  const int n = n0 + l15;
  const float bv = bias[n];
#pragma unroll
  for (int r = 0; r < 8; ++r) {
    const int m = m0 + r + 8 * half;   // C-layout: VGPR r -> row r (+8 for upper half)
    float v = acc[r] + bv;
    if (mode == 3) {
      ((float*)dstv)[(size_t)m * Nc + n] = v;
    } else {
      const int b = m >> 11, s = m & (Sc - 1);
      const int h = n >> 6, d = n & (HDc - 1);
      bf16* dst = (bf16*)dstv;
      if (mode == 0)
        dst[((size_t)(b * Hc + h) * Sc + s) * HDc + d] = (bf16)(v * QSCALE);
      else if (mode == 1)
        dst[((size_t)(b * Hc + h) * Sc + s) * HDc + d] = (bf16)v;
      else
        dst[((size_t)(b * Hc + h) * HDc + d) * Sc + s] = (bf16)v;
    }
  }
}

__global__ __launch_bounds__(256) void k_gemm(const bf16* __restrict__ A,
                                              const bf16* __restrict__ Bt,
                                              const float* __restrict__ bias,
                                              void* __restrict__ dstv, int mode) {
  const int lane = threadIdx.x & 31;
  const int wave = threadIdx.x >> 5;  // 0..7 -> 4 (M) x 2 (N)
  const int half = lane >> 4;
  const int l15 = lane & 15;
  const int mBase = blockIdx.x * 128 + (wave & 3) * 32;
  const int nBase = blockIdx.y * 64 + (wave >> 2) * 32;

  const bf16* a0p = A + (size_t)(mBase + l15) * Dc;
  const bf16* a1p = A + (size_t)(mBase + 16 + l15) * Dc;
  const bf16* b0p = Bt + (size_t)(nBase + l15) * Dc;
  const bf16* b1p = Bt + (size_t)(nBase + 16 + l15) * Dc;

  v8f acc00 = (v8f)0.f, acc01 = (v8f)0.f, acc10 = (v8f)0.f, acc11 = (v8f)0.f;
#pragma unroll 2
  for (int kc = 0; kc < Dc; kc += 32) {
    __builtin_prefetch(a0p + kc + 128, 0, 3);
    __builtin_prefetch(b0p + kc + 128, 0, 3);
    v16bf a0 = load_a_frag(a0p, kc, half);
    v16bf a1 = load_a_frag(a1p, kc, half);
    v16bf b0 = load_b_frag(b0p, kc, half);
    v16bf b1 = load_b_frag(b1p, kc, half);
    acc00 = wmma_bf16(a0, b0, acc00);
    acc01 = wmma_bf16(a0, b1, acc01);
    acc10 = wmma_bf16(a1, b0, acc10);
    acc11 = wmma_bf16(a1, b1, acc11);
  }
  store_tile(acc00, mBase, nBase, l15, half, bias, dstv, mode);
  store_tile(acc01, mBase, nBase + 16, l15, half, bias, dstv, mode);
  store_tile(acc10, mBase + 16, nBase, l15, half, bias, dstv, mode);
  store_tile(acc11, mBase + 16, nBase + 16, l15, half, bias, dstv, mode);
}

// ---------------- Flash attention ----------------
// grid: (S/128, B*H), block 256 (8 waves). One wave = 16 query rows.
// K/V 32-key chunks staged in LDS via TDM (wave 0, TENSORcnt) or async copies,
// shared by all 8 waves.
__global__ __launch_bounds__(256) void k_attn(const bf16* __restrict__ Q,
                                              const bf16* __restrict__ K,
                                              const bf16* __restrict__ Vt,
                                              bf16* __restrict__ ctx) {
  __shared__ __align__(32) bf16 Kt[32][64];      // [key-in-chunk][hd]      4 KB
  __shared__ __align__(32) bf16 Vtile[64][32];   // [hd][key-in-chunk]      4 KB
  __shared__ __align__(32) bf16 pst[8][16][32];  // per-wave P staging      8 KB
  const int tid = threadIdx.x;
  const int lane = tid & 31;
  const int wave = tid >> 5;                     // 0..7
  const int half = lane >> 4;
  const int l15 = lane & 15;
  const int bh = blockIdx.y;                     // b*H + h
  const int qBase = blockIdx.x * 128 + wave * 16;

  const bf16* Qh = Q + (size_t)bh * Sc * HDc;
  const bf16* Kh = K + (size_t)bh * Sc * HDc;
  const bf16* Vh = Vt + (size_t)bh * HDc * Sc;

  const bf16* qrow = Qh + (size_t)(qBase + l15) * HDc;
  const v16bf qf0 = load_a_frag(qrow, 0, half);   // hd 0..31
  const v16bf qf1 = load_a_frag(qrow, 32, half);  // hd 32..63

  v8f o0 = (v8f)0.f, o1 = (v8f)0.f, o2 = (v8f)0.f, o3 = (v8f)0.f;
  float mrun[8], lrun[8], alpha[8];
#pragma unroll
  for (int r = 0; r < 8; ++r) { mrun[r] = -3.0e38f; lrun[r] = 0.f; }

  bf16* myp = &pst[wave][0][0];
  // fallback cooperative staging mapping: 256 threads x 16B = one 4KB tile
  const int krow = tid >> 3, kseg = tid & 7;     // K tile: 32 rows x 8 segs
  const int vrow = tid >> 2, vseg = tid & 3;     // V tile: 64 rows x 4 segs

  for (int kb = 0; kb < Sc; kb += 32) {
    if (kb != 0) __syncthreads();  // previous tile fully consumed
#if HAVE_TDM
    if (wave == 0) {
      // K chunk: 32 rows x 64 cols, row stride 64; V chunk: 64 rows x 32 cols, stride Sc
      tdm_load_tile(Kh + (size_t)kb * HDc, &Kt[0][0], 32, 64, (unsigned long long)HDc);
      tdm_load_tile(Vh + kb, &Vtile[0][0], 64, 32, (unsigned long long)Sc);
      tdm_wait_all();
    }
#else
    async_copy16(Kh + (size_t)(kb + krow) * HDc + kseg * 8, &Kt[krow][kseg * 8]);
    async_copy16(Vh + (size_t)vrow * Sc + kb + vseg * 8, &Vtile[vrow][vseg * 8]);
    async_wait_all();
#endif
    __syncthreads();               // tile visible to all waves

    // scores: Q(16x64) x K^T(64x32) from LDS
    const bf16* k0 = &Kt[l15][0];
    const bf16* k1 = &Kt[16 + l15][0];
    v8f s0 = (v8f)0.f, s1 = (v8f)0.f;
    s0 = wmma_bf16(qf0, load_b_frag(k0, 0, half), s0);
    s0 = wmma_bf16(qf1, load_b_frag(k0, 32, half), s0);
    s1 = wmma_bf16(qf0, load_b_frag(k1, 0, half), s1);
    s1 = wmma_bf16(qf1, load_b_frag(k1, 32, half), s1);

    // online softmax per row (rows live in 16-lane halves; stats per VGPR slot r)
#pragma unroll
    for (int r = 0; r < 8; ++r) {
      float cm = redmax16(fmaxf(s0[r], s1[r]));
      float mn = fmaxf(mrun[r], cm);
      alpha[r] = __builtin_amdgcn_exp2f(mrun[r] - mn);
      mrun[r] = mn;
      float p0 = __builtin_amdgcn_exp2f(s0[r] - mn);
      float p1 = __builtin_amdgcn_exp2f(s1[r] - mn);
      s0[r] = p0;
      s1[r] = p1;
      lrun[r] = lrun[r] * alpha[r] + redsum16(p0 + p1);
    }

    // C-layout P -> LDS (row-major 16x32), same-wave only: no barrier needed
#pragma unroll
    for (int r = 0; r < 8; ++r) {
      const int row = r + 8 * half;
      myp[row * 32 + l15] = (bf16)s0[r];
      myp[row * 32 + 16 + l15] = (bf16)s1[r];
    }
    // read back in A-fragment layout: row = l15, two 16B chunks
    union { v16bf v; v8bf h[2]; } pu;
    pu.h[0] = *(const v8bf*)(myp + l15 * 32 + 8 * half);
    pu.h[1] = *(const v8bf*)(myp + l15 * 32 + 16 + 8 * half);

    // rescale O and accumulate P*V over this 32-key chunk (4 hd-tiles of 16)
#pragma unroll
    for (int r = 0; r < 8; ++r) {
      o0[r] *= alpha[r]; o1[r] *= alpha[r]; o2[r] *= alpha[r]; o3[r] *= alpha[r];
    }
    const bf16* vt0 = &Vtile[0 * 16 + l15][0];
    const bf16* vt1 = &Vtile[1 * 16 + l15][0];
    const bf16* vt2 = &Vtile[2 * 16 + l15][0];
    const bf16* vt3 = &Vtile[3 * 16 + l15][0];
    o0 = wmma_bf16(pu.v, load_b_frag(vt0, 0, half), o0);
    o1 = wmma_bf16(pu.v, load_b_frag(vt1, 0, half), o1);
    o2 = wmma_bf16(pu.v, load_b_frag(vt2, 0, half), o2);
    o3 = wmma_bf16(pu.v, load_b_frag(vt3, 0, half), o3);
  }

  // normalize and store ctx bf16 [B,S,H*HD]
  const int h = bh & (Hc - 1), b = bh >> 4;
#pragma unroll
  for (int r = 0; r < 8; ++r) {
    const int q = qBase + r + 8 * half;
    const float inv = 1.0f / lrun[r];
    bf16* dst = ctx + ((size_t)b * Sc + q) * Nc + h * HDc + l15;
    dst[0 * 16] = (bf16)(o0[r] * inv);
    dst[1 * 16] = (bf16)(o1[r] * inv);
    dst[2 * 16] = (bf16)(o2[r] * inv);
    dst[3 * 16] = (bf16)(o3[r] * inv);
  }
}

// ---------------- host launcher ----------------
extern "C" void kernel_launch(void* const* d_in, const int* in_sizes, int n_in,
                              void* d_out, int out_size, void* d_ws, size_t ws_size,
                              hipStream_t stream) {
  (void)in_sizes; (void)n_in; (void)out_size; (void)ws_size;
  const float* x  = (const float*)d_in[0];
  const float* wq = (const float*)d_in[1];
  const float* bq = (const float*)d_in[2];
  const float* wk = (const float*)d_in[3];
  const float* bk = (const float*)d_in[4];
  const float* wv = (const float*)d_in[5];
  const float* bv = (const float*)d_in[6];
  const float* wo = (const float*)d_in[7];
  const float* bo = (const float*)d_in[8];

  char* w = (char*)d_ws;
  bf16* xbf = (bf16*)(w);                     // 8 MB  [4096,1024]
  bf16* wqT = (bf16*)(w + (8ull << 20));      // 2 MB  [1024(n),1024(k)]
  bf16* wkT = (bf16*)(w + (10ull << 20));     // 2 MB
  bf16* wvT = (bf16*)(w + (12ull << 20));     // 2 MB
  bf16* woT = (bf16*)(w + (14ull << 20));     // 2 MB
  bf16* Qw  = (bf16*)(w + (16ull << 20));     // 8 MB  [B,H,S,HD]
  bf16* Kw  = (bf16*)(w + (24ull << 20));     // 8 MB  [B,H,S,HD]
  bf16* Vtw = (bf16*)(w + (32ull << 20));     // 8 MB  [B,H,HD,S]
  bf16* ctx = (bf16*)(w + (40ull << 20));     // 8 MB  [4096,1024]  (total 48 MB)

  k_cast_bf16<<<(Mc * Dc) / 256, 256, 0, stream>>>(x, xbf, Mc * Dc);
  k_transpose_cast<<<(Dc * Nc) / 256, 256, 0, stream>>>(wq, wqT);
  k_transpose_cast<<<(Dc * Nc) / 256, 256, 0, stream>>>(wk, wkT);
  k_transpose_cast<<<(Dc * Nc) / 256, 256, 0, stream>>>(wv, wvT);
  k_transpose_cast<<<(Dc * Nc) / 256, 256, 0, stream>>>(wo, woT);

  dim3 gg(Mc / 128, Nc / 64);
  k_gemm<<<gg, 256, 0, stream>>>(xbf, wqT, bq, (void*)Qw, 0);
  k_gemm<<<gg, 256, 0, stream>>>(xbf, wkT, bk, (void*)Kw, 1);
  k_gemm<<<gg, 256, 0, stream>>>(xbf, wvT, bv, (void*)Vtw, 2);

  k_attn<<<dim3(Sc / 128, Bc * Hc), 256, 0, stream>>>(Qw, Kw, Vtw, ctx);

  k_gemm<<<gg, 256, 0, stream>>>(ctx, woT, bo, d_out, 3);
}
